// Controller_19018115187142
// MI455X (gfx1250) — matrix operand order repaired and checked by reading, not compiled, not run on hardware
//
#include <hip/hip_runtime.h>
#include <cstdint>
#include <cstddef>

// ---------------------------------------------------------------------------
// 2-layer LSTM forecaster for gfx1250 (MI455X).
// Batch-parallel persistent blocks: each block owns 32 batch rows, keeps cell
// state in VGPRs, hidden state + weights in LDS, drives all GEMMs through
// v_wmma_f32_16x16x32_bf16, and stages weights / h0 tiles with the Tensor
// Data Mover when the builtin is available.
// ---------------------------------------------------------------------------

typedef unsigned short u16;
typedef __attribute__((ext_vector_type(16))) __bf16 v16bf;
typedef __attribute__((ext_vector_type(8)))  __bf16 v8bf;
typedef __attribute__((ext_vector_type(8)))  float  v8f;
typedef __attribute__((ext_vector_type(4)))  unsigned int u32x4;
typedef __attribute__((ext_vector_type(8)))  int i32x8;
typedef __attribute__((ext_vector_type(4)))  int i32x4;

static constexpr int BATCH = 8192;
static constexpr int TLB   = 48;     // lookback
static constexpr int THOR  = 12;     // horizon
static constexpr int TTOT  = TLB + THOR;
static constexpr int DIN   = 8;
static constexpr int H     = 128;
static constexpr int G4    = 512;    // 4*H gate columns
static constexpr int BT    = 32;     // batch rows per block
static constexpr int NBLK  = BATCH / BT;
static constexpr int K0S   = 168;    // layer0 K stride: 128 h + 8 x + pad (bank-conflict-free)
static constexpr int K1S   = 264;    // layer1 K stride: 128 h0 + 128 h1 + pad
static constexpr int KT0   = 5;      // k-tiles of 32 for layer0 (covers 0..159)
static constexpr int KT1   = 8;      // k-tiles of 32 for layer1 (covers 0..255)

// workspace layout (bytes)
static constexpr size_t WS_W0 = 0;
static constexpr size_t SZ_W0 = (size_t)G4 * K0S * 2;           // 172032
static constexpr size_t WS_W1 = WS_W0 + SZ_W0;
static constexpr size_t SZ_W1 = (size_t)G4 * K1S * 2;           // 270336
static constexpr size_t WS_B0 = WS_W1 + SZ_W1;
static constexpr size_t WS_B1 = WS_B0 + (size_t)G4 * 4;
static constexpr size_t WS_H0 = WS_B1 + (size_t)G4 * 4;         // h0 stream [48][8192][128] bf16
static constexpr size_t SZ_H0 = (size_t)TLB * BATCH * H * 2;    // ~100 MB (L2-resident)

// LDS layout (bytes)
static constexpr unsigned L_W     = 0;                           // weight matrix (max 270336)
static constexpr unsigned L_A0    = 270336;                      // layer0 A rows  [BT][K0S] bf16
static constexpr unsigned L_A1    = L_A0 + BT * K0S * 2;         // layer1 A rows  [BT][K1S] bf16
static constexpr unsigned L_PRED  = L_A1 + BT * K1S * 2;         // pred feedback  [BT] f32
static constexpr unsigned LDS_TOT = L_PRED + BT * 4;             // 298112 <= 320KB

#if defined(__HIP_DEVICE_COMPILE__)
#  if __has_builtin(__builtin_amdgcn_tensor_load_to_lds)
#    define HAVE_TDM 1
#  else
#    define HAVE_TDM 0
#  endif
#else
#  define HAVE_TDM 0
#endif

// ---------------- small numeric helpers ----------------
__device__ __forceinline__ u16 f2bf(float f) {
  unsigned u = __builtin_bit_cast(unsigned, f);
  unsigned r = u + 0x7fffu + ((u >> 16) & 1u);   // RNE
  return (u16)(r >> 16);
}
__device__ __forceinline__ float bf2f(u16 h) {
  return __builtin_bit_cast(float, (unsigned)h << 16);
}
__device__ __forceinline__ float sigf(float x) {
  return __builtin_amdgcn_rcpf(1.0f + __expf(-x));
}
__device__ __forceinline__ float tanh_fast(float x) {
  float e = __expf(2.0f * x);                    // inf/0 saturate correctly
  return 1.0f - 2.0f * __builtin_amdgcn_rcpf(e + 1.0f);
}

// ---------------- WMMA fragment loaders ----------------
__device__ __forceinline__ v16bf cat8(v8bf lo, v8bf hi) {
  v16bf r;
#pragma unroll
  for (int i = 0; i < 8; ++i) { r[i] = lo[i]; r[i + 8] = hi[i]; }
  return r;
}
// A (16x32 bf16): lane m=l&15; lanes<16 hold K {0..7,16..23}, lanes>=16 {8..15,24..31}
__device__ __forceinline__ v16bf ld_fragA(const u16* base, int stride, int rbase, int kt, int lane) {
  const u16* p = base + (rbase + (lane & 15)) * stride + kt * 32 + ((lane >> 4) << 3);
  return cat8(*(const v8bf*)p, *(const v8bf*)(p + 16));
}
// B (32x16 bf16): lane n=l&15; lanes<16 hold K {0..7,16..23}, lanes>=16 {8..15,24..31}
__device__ __forceinline__ v16bf ld_fragB(const u16* base, int stride, int nbase, int kt, int lane) {
  const u16* p = base + (nbase + (lane & 15)) * stride + kt * 32 + ((lane >> 4) << 3);
  return cat8(*(const v8bf*)p, *(const v8bf*)(p + 16));
}

// 4-gate GEMM: acc[mtile][gate] += A[BT x 32k] * W[32k x 16]  over KT k-tiles
template <int KT>
__device__ __forceinline__ void lstm_gemm(const u16* __restrict__ A, int astride,
                                          const u16* __restrict__ W, int wstride,
                                          int j, int lane, v8f (&acc)[2][4]) {
#pragma unroll
  for (int kt = 0; kt < KT; ++kt) {
    v16bf a0 = ld_fragA(A, astride, 0, kt, lane);
    v16bf a1 = ld_fragA(A, astride, 16, kt, lane);
#pragma unroll
    for (int g = 0; g < 4; ++g) {
      v16bf b = ld_fragB(W, wstride, g * H + j, kt, lane);
      acc[0][g] = __builtin_amdgcn_wmma_f32_16x16x32_bf16(false, a0, false, b, (short)0, acc[0][g], false, false);
      acc[1][g] = __builtin_amdgcn_wmma_f32_16x16x32_bf16(false, a1, false, b, (short)0, acc[1][g], false, false);
    }
  }
}

// ---------------- Tensor Data Mover (2D tile -> LDS, optional row padding) ----
#if HAVE_TDM
__device__ __forceinline__ void tdm_load2d(unsigned lds_byte_off, const void* gptr,
                                           unsigned tileX, unsigned tileY,
                                           unsigned long long rowStrideElems,
                                           unsigned tensorX, unsigned tensorY,
                                           unsigned padIntervalCode, unsigned padAmountCode,
                                           bool padEn) {
  unsigned long long ga = (unsigned long long)(uintptr_t)gptr;
  u32x4 g0;
  g0[0] = 1u;                                             // count=1, user descriptor
  g0[1] = lds_byte_off;                                   // lds_addr
  g0[2] = (unsigned)ga;                                   // global_addr[31:0]
  g0[3] = (unsigned)((ga >> 32) & 0x01ffffffu) | (2u << 30); // addr[56:32] | type=2
  // group1 bitfields (see CDNA5 ISA)
  unsigned long long w0 = (1ull << 16)                    // data_size = 2 bytes (bf16)
      | (padEn ? ((1ull << 20) | ((unsigned long long)padIntervalCode << 22)
                               | ((unsigned long long)padAmountCode << 25)) : 0ull)
      | ((unsigned long long)(tensorX & 0xffffu) << 48);  // tensor_dim0[15:0]
  unsigned long long w1 = (unsigned long long)(tensorX >> 16)
      | ((unsigned long long)tensorY << 16)               // tensor_dim1
      | ((unsigned long long)(tileX & 0xffffu) << 48);    // tile_dim0
  unsigned long long w2 = (unsigned long long)(tileY & 0xffffu)   // tile_dim1
      | ((rowStrideElems & 0xffffffffull) << 32);         // tensor_dim0_stride lo
  unsigned long long w3 = ((rowStrideElems >> 32) & 0xffffull);
  i32x8 g1;
  g1[0] = (int)w0; g1[1] = (int)(w0 >> 32);
  g1[2] = (int)w1; g1[3] = (int)(w1 >> 32);
  g1[4] = (int)w2; g1[5] = (int)(w2 >> 32);
  g1[6] = (int)w3; g1[7] = (int)(w3 >> 32);
  i32x4 z4 = {0, 0, 0, 0};
#if __clang_major__ >= 23
  i32x8 z8 = {0, 0, 0, 0, 0, 0, 0, 0};
  __builtin_amdgcn_tensor_load_to_lds(g0, g1, z4, z4, z8, 0);
#else
  __builtin_amdgcn_tensor_load_to_lds(g0, g1, z4, z4, 0);
#endif
}
#endif

// ---------------- prep kernel: pack bf16 combined weights + fused biases -----
__global__ void lstm_prep_kernel(const float* __restrict__ Wih0, const float* __restrict__ Whh0,
                                 const float* __restrict__ bih0, const float* __restrict__ bhh0,
                                 const float* __restrict__ Wih1, const float* __restrict__ Whh1,
                                 const float* __restrict__ bih1, const float* __restrict__ bhh1,
                                 char* __restrict__ ws) {
  int tid = blockIdx.x * blockDim.x + threadIdx.x;
  u16* w0 = (u16*)(ws + WS_W0);
  u16* w1 = (u16*)(ws + WS_W1);
  float* b0 = (float*)(ws + WS_B0);
  float* b1 = (float*)(ws + WS_B1);
  const int n0 = G4 * K0S, n1 = G4 * K1S;
  if (tid < n0) {                       // Wcat0[n][k] = k<128 ? Whh0 : k<136 ? Wih0 : 0
    int n = tid / K0S, k = tid % K0S;
    float v = 0.f;
    if (k < H) v = Whh0[n * H + k];
    else if (k < H + DIN) v = Wih0[n * DIN + (k - H)];
    w0[tid] = f2bf(v);
  } else if (tid < n0 + n1) {           // Wcat1[n][k] = k<128 ? Wih1 : k<256 ? Whh1 : 0
    int t = tid - n0;
    int n = t / K1S, k = t % K1S;
    float v = 0.f;
    if (k < H) v = Wih1[n * H + k];
    else if (k < 2 * H) v = Whh1[n * H + (k - H)];
    w1[t] = f2bf(v);
  } else if (tid < n0 + n1 + G4) {
    int n = tid - n0 - n1;
    b0[n] = bih0[n] + bhh0[n];
  } else if (tid < n0 + n1 + 2 * G4) {
    int n = tid - n0 - n1 - G4;
    b1[n] = bih1[n] + bhh1[n];
  }
}

// ---------------- main persistent LSTM kernel --------------------------------
__global__ __launch_bounds__(256, 1)
void lstm_main_kernel(const float* __restrict__ inputs, const float* __restrict__ Wfc,
                      const float* __restrict__ bfc, char* __restrict__ ws,
                      float* __restrict__ out) {
  extern __shared__ char smem[];
  u16*   Wsm   = (u16*)(smem + L_W);
  u16*   A0    = (u16*)(smem + L_A0);
  u16*   A1    = (u16*)(smem + L_A1);
  float* predb = (float*)(smem + L_PRED);

  const int tid   = threadIdx.x;
  const int lane  = tid & 31;
  const int wid   = tid >> 5;
  const int j     = wid * 16;          // this wave's gate-column slice within H
  const int Bbase = blockIdx.x * BT;

  const u16*   W0g = (const u16*)(ws + WS_W0);
  const u16*   W1g = (const u16*)(ws + WS_W1);
  const float* b0g = (const float*)(ws + WS_B0);
  const float* b1g = (const float*)(ws + WS_B1);
  u16*         h0s = (u16*)(ws + WS_H0);

  float bias0[4], bias1[4];
#pragma unroll
  for (int g = 0; g < 4; ++g) {
    bias0[g] = b0g[g * H + j + (lane & 15)];
    bias1[g] = b1g[g * H + j + (lane & 15)];
  }

  // zero A buffers (h=0 init, zero K-padding regions)
  for (int i = tid; i < (BT * K0S) / 2; i += 256) ((unsigned*)A0)[i] = 0u;
  for (int i = tid; i < (BT * K1S) / 2; i += 256) ((unsigned*)A1)[i] = 0u;

  v8f c0[2], c1[2];
#pragma unroll
  for (int mt = 0; mt < 2; ++mt)
#pragma unroll
    for (int e = 0; e < 8; ++e) { c0[mt][e] = 0.f; c1[mt][e] = 0.f; }

  v8f acc[2][4];
  auto init_acc = [&](const float* bv) {
#pragma unroll
    for (int mt = 0; mt < 2; ++mt)
#pragma unroll
      for (int g = 0; g < 4; ++g) {
        v8f t;
#pragma unroll
        for (int e = 0; e < 8; ++e) t[e] = bv[g];
        acc[mt][g] = t;
      }
  };
  auto gates = [&](v8f (&c)[2], auto&& wr) {
#pragma unroll
    for (int mt = 0; mt < 2; ++mt) {
#pragma unroll
      for (int e = 0; e < 8; ++e) {
        float gi = acc[mt][0][e], gf = acc[mt][1][e];
        float gg = acc[mt][2][e], go = acc[mt][3][e];
        float cn = sigf(gf) * c[mt][e] + sigf(gi) * tanh_fast(gg);
        float hh = sigf(go) * tanh_fast(cn);
        c[mt][e] = cn;
        int row = mt * 16 + ((lane >> 4) << 3) + e;   // D-matrix VGPR layout
        int col = j + (lane & 15);
        wr(row, col, f2bf(hh));
      }
    }
  };

  // ---- stage layer-0 combined weights into LDS ----
#if HAVE_TDM
  if (wid == 0) {
    tdm_load2d(L_W, W0g, K0S, G4, K0S, K0S, G4, 0, 0, false);
    __builtin_amdgcn_s_wait_tensorcnt(0);
  }
#else
  for (size_t i = tid; i < SZ_W0 / 16; i += 256)
    ((uint4*)Wsm)[i] = ((const uint4*)W0g)[i];
#endif
  __syncthreads();

  // =================== Phase 1: encoder, layer 0 (t = 0..47) ===================
  for (int t = 0; t < TLB; ++t) {
    { // stage x_t into K columns 128..135
      int b = tid >> 3, d = tid & 7;
      A0[b * K0S + H + d] = f2bf(inputs[(size_t)(Bbase + b) * TTOT * DIN + t * DIN + d]);
    }
    __syncthreads();
    init_acc(bias0);
    lstm_gemm<KT0>(A0, K0S, Wsm, K0S, j, lane, acc);
    __syncthreads();
    gates(c0, [&](int r, int c, u16 hb) {
      A0[r * K0S + c] = hb;                                          // recurrent h0
      h0s[((size_t)t * BATCH + Bbase + r) * H + c] = hb;             // stream for layer 1
    });
  }
  __syncthreads();

  // ---- swap in layer-1 combined weights ----
#if HAVE_TDM
  if (wid == 0) {
    tdm_load2d(L_W, W1g, K1S, G4, K1S, K1S, G4, 0, 0, false);
    __builtin_amdgcn_s_wait_tensorcnt(0);
  }
#else
  for (size_t i = tid; i < SZ_W1 / 16; i += 256)
    ((uint4*)Wsm)[i] = ((const uint4*)W1g)[i];
#endif
  __syncthreads();

  // =================== Phase 2: encoder, layer 1 (t = 0..47) ===================
  for (int t = 0; t < TLB; ++t) {
#if HAVE_TDM
    if (wid == 0) {
      // 32x128 bf16 tile; TDM pads each 256B row out to the 528B LDS stride:
      // pad after 64 DWORDs (code 5), pad 68 DWORDs (code 67)
      tdm_load2d(L_A1, h0s + ((size_t)t * BATCH + Bbase) * H, H, BT, H, H, BT, 5, 67, true);
      __builtin_amdgcn_s_wait_tensorcnt(0);
    }
#else
    { int b = tid >> 3, seg = tid & 7;
      const uint4* s = (const uint4*)(h0s + ((size_t)t * BATCH + Bbase + b) * H + seg * 16);
      uint4* d = (uint4*)(A1 + b * K1S + seg * 16);
      d[0] = s[0]; d[1] = s[1]; }
#endif
    __syncthreads();
    init_acc(bias1);
    lstm_gemm<KT1>(A1, K1S, Wsm, K1S, j, lane, acc);
    __syncthreads();
    gates(c1, [&](int r, int c, u16 hb) { A1[r * K1S + H + c] = hb; });
  }
  __syncthreads();

  // =================== Phase 3: autoregressive decoder (t = 0..11) ============
  for (int t = 0; t < THOR; ++t) {
    { // stage x with blood-glucose feedback (channel 0)
      int b = tid >> 3, d = tid & 7;
      float v = inputs[(size_t)(Bbase + b) * TTOT * DIN + (TLB + t) * DIN + d];
      if (d == 0 && t > 0) v = predb[b];
      A0[b * K0S + H + d] = f2bf(v);
    }
    __syncthreads();
    // layer 0: weights streamed from global (L2-resident), LDS holds layer-1 W
    init_acc(bias0);
    lstm_gemm<KT0>(A0, K0S, W0g, K0S, j, lane, acc);
    __syncthreads();
    gates(c0, [&](int r, int c, u16 hb) {
      A0[r * K0S + c] = hb;          // recurrent h0
      A1[r * K1S + c] = hb;          // feeds layer 1 this step
    });
    __syncthreads();
    // layer 1: LDS weights
    init_acc(bias1);
    lstm_gemm<KT1>(A1, K1S, Wsm, K1S, j, lane, acc);
    __syncthreads();
    gates(c1, [&](int r, int c, u16 hb) { A1[r * K1S + H + c] = hb; });
    __syncthreads();
    // FC head: pred[b] = h1[b,:] . Wfc + bfc ; write output + feedback
    if (wid == 0) {
      float s = bfc[0];
      for (int k = 0; k < H; ++k) s += bf2f(A1[lane * K1S + H + k]) * Wfc[k];
      predb[lane] = s;
      out[(size_t)(Bbase + lane) * THOR + t] = s;
    }
    __syncthreads();
  }
}

// ---------------------------------------------------------------------------
extern "C" void kernel_launch(void* const* d_in, const int* in_sizes, int n_in,
                              void* d_out, int out_size, void* d_ws, size_t ws_size,
                              hipStream_t stream) {
  (void)in_sizes; (void)n_in; (void)out_size; (void)ws_size;
  const float* inputs = (const float*)d_in[0];
  const float* Wih0   = (const float*)d_in[1];
  const float* Whh0   = (const float*)d_in[2];
  const float* bih0   = (const float*)d_in[3];
  const float* bhh0   = (const float*)d_in[4];
  const float* Wih1   = (const float*)d_in[5];
  const float* Whh1   = (const float*)d_in[6];
  const float* bih1   = (const float*)d_in[7];
  const float* bhh1   = (const float*)d_in[8];
  const float* Wfc    = (const float*)d_in[9];
  const float* bfc    = (const float*)d_in[10];
  char*  ws  = (char*)d_ws;
  float* out = (float*)d_out;

  // allow >64KB dynamic LDS (idempotent; errors ignored)
  (void)hipFuncSetAttribute(reinterpret_cast<const void*>(lstm_main_kernel),
                            hipFuncAttributeMaxDynamicSharedMemorySize, (int)LDS_TOT);

  const int prep_threads = G4 * K0S + G4 * K1S + 2 * G4;
  lstm_prep_kernel<<<(prep_threads + 255) / 256, 256, 0, stream>>>(
      Wih0, Whh0, bih0, bhh0, Wih1, Whh1, bih1, bhh1, ws);
  lstm_main_kernel<<<NBLK, 256, LDS_TOT, stream>>>(inputs, Wfc, bfc, ws, out);
}
